// HAN_20323785244854
// MI455X (gfx1250) — compile-verified, matrix operand order checked by reading, not run on hardware
//
#include <hip/hip_runtime.h>
#include <hip/hip_bf16.h>
#include <math.h>
#include <stdint.h>

// ---------------------------------------------------------------------------
// HANConv (author/paper) for gfx1250.
//  - f16 WMMA GEMMs (v_wmma_f32_16x16x32_f16), fp32 accumulate
//  - B-matrix pre-packed in fragment order, staged to LDS via the Tensor
//    Data Mover (tensor_load_to_lds + s_wait_tensorcnt), read as ds_load_b128
//  - edge softmax/scatter with float atomics (node tables are L2-resident)
// ---------------------------------------------------------------------------

typedef __attribute__((ext_vector_type(16))) _Float16 v16h;
typedef __attribute__((ext_vector_type(8)))  float    v8f;
typedef __attribute__((ext_vector_type(4)))  float    v4f;
typedef __attribute__((ext_vector_type(4)))  unsigned int v4u;
typedef __attribute__((ext_vector_type(8)))  int      v8i_;
typedef __attribute__((ext_vector_type(4)))  int      v4i_;

#define HID 128
#define NHEAD 8
#define DHEAD 16
#define OUTC 64

#if defined(__has_builtin)
#if __has_builtin(__builtin_amdgcn_tensor_load_to_lds)
#define HAVE_TDM 1
#endif
#endif
#ifndef HAVE_TDM
#define HAVE_TDM 0
#endif

// K index inside a 16-half WMMA fragment (ISA 7.12.2, mirrored for A and B):
__device__ __forceinline__ int kmap(int h, int lane) {
    return h + (h & 8) + ((lane & 16) >> 1);
}

// ---- A fragment: the kmap pattern is two contiguous 8-float runs ----------
__device__ __forceinline__ v16h cvt16(v4f q0, v4f q1, v4f q2, v4f q3) {
    v16h f;
#pragma unroll
    for (int j = 0; j < 4; ++j) {
        f[j]      = (_Float16)q0[j];
        f[j + 4]  = (_Float16)q1[j];
        f[j + 8]  = (_Float16)q2[j];
        f[j + 12] = (_Float16)q3[j];
    }
    return f;
}

__device__ __forceinline__ v16h load_frag_a(const float* __restrict__ X, int ld,
                                            int rowBase, int kBase, int lane) {
    const float* p = X + (size_t)(rowBase + (lane & 15)) * ld + kBase + ((lane & 16) >> 1);
    return cvt16(*(const v4f*)p, *(const v4f*)(p + 4),
                 *(const v4f*)(p + 16), *(const v4f*)(p + 20));
}

__device__ __forceinline__ v16h load_frag_a_affine(const float* __restrict__ X, int ld,
                                                   int rowBase, int kBase, int lane,
                                                   const float* __restrict__ ga,
                                                   const float* __restrict__ gb) {
    const int o = kBase + ((lane & 16) >> 1);
    const float* p = X + (size_t)(rowBase + (lane & 15)) * ld + o;
    v4f q0 = *(const v4f*)(p)      * *(const v4f*)(ga + o)      + *(const v4f*)(gb + o);
    v4f q1 = *(const v4f*)(p + 4)  * *(const v4f*)(ga + o + 4)  + *(const v4f*)(gb + o + 4);
    v4f q2 = *(const v4f*)(p + 16) * *(const v4f*)(ga + o + 16) + *(const v4f*)(gb + o + 16);
    v4f q3 = *(const v4f*)(p + 20) * *(const v4f*)(ga + o + 20) + *(const v4f*)(gb + o + 20);
    return cvt16(q0, q1, q2, q3);
}

__device__ __forceinline__ v8f wmma16(v16h a, v16h b, v8f c) {
    return __builtin_amdgcn_wmma_f32_16x16x32_f16(false, a, false, b, (short)0, c,
                                                  false, false);
}

// ---- TDM: 1-D dword copy global -> LDS (D# per ISA 8.3-8.6) ---------------
__device__ __forceinline__ void tdm_load_to_lds(const void* gsrc, unsigned ldsOff,
                                                unsigned ndwords) {
#if HAVE_TDM
    unsigned long long ga = (unsigned long long)(uintptr_t)gsrc;
    v4u g0;
    g0[0] = 1u;                                   // count=1, user descriptor
    g0[1] = ldsOff;                               // lds_addr
    g0[2] = (unsigned)(ga & 0xffffffffu);         // global_addr[31:0]
    g0[3] = (unsigned)((ga >> 32) & 0x01ffffffu)  // global_addr[56:32]
          | (2u << 30);                           // type=2 ("image")
    v8i_ g1;
    g1[0] = (int)(2u << 16);                      // wg_mask=0, data_size=2 (4B)
    g1[1] = (int)((ndwords & 0xffffu) << 16);     // tensor_dim0[15:0]
    g1[2] = (int)(((ndwords >> 16) & 0xffffu)     // tensor_dim0[31:16]
          | (1u << 16));                          // tensor_dim1 = 1
    g1[3] = (int)((ndwords & 0xffffu) << 16);     // tile_dim0
    g1[4] = 0;                                    // tile_dim1=0, tile_dim2=0
    g1[5] = (int)ndwords;                         // tensor_dim0_stride[31:0]
    g1[6] = 0;
    g1[7] = 0;
    v4i_ z4 = {0, 0, 0, 0};
    v8i_ z8 = {0, 0, 0, 0, 0, 0, 0, 0};
    __builtin_amdgcn_tensor_load_to_lds(g0, g1, z4, z4, z8, 0);
#else
    (void)gsrc; (void)ldsOff; (void)ndwords;
#endif
}

// stage NBYTES of fragment-packed f16 weights into LDS (whole block)
template <int NBYTES>
__device__ __forceinline__ void stage_w(const _Float16* __restrict__ gW,
                                        _Float16* sm) {
#if HAVE_TDM
    if (threadIdx.x < 32) {                       // wave 0 issues the DMA
        tdm_load_to_lds((const void*)gW, (unsigned)(uintptr_t)(void*)sm, NBYTES / 4);
        __builtin_amdgcn_s_wait_tensorcnt(0);
    }
    __syncthreads();
#else
    const uint4* src = (const uint4*)gW;
    uint4* dst = (uint4*)sm;
    for (int i = threadIdx.x; i < NBYTES / 16; i += blockDim.x) dst[i] = src[i];
    __syncthreads();
#endif
}

// pack W (f32, row-major [K, nCols]) into f16 fragment order:
// out[((ct*4+kb)*32 + lane)*16 + h] = W[kb*32 + kmap(h,lane), ct*16 + lane%16]
__global__ void pack_wfrag_kernel(const float* __restrict__ W, _Float16* __restrict__ out,
                                  int nCols) {
    int i = blockIdx.x * blockDim.x + threadIdx.x;
    int total = (nCols >> 4) * 2048;
    if (i >= total) return;
    int h = i & 15;
    int lane = (i >> 4) & 31;
    int kb = (i >> 9) & 3;
    int ct = i >> 11;
    int K = kb * 32 + kmap(h, lane);
    int N = ct * 16 + (lane & 15);
    out[i] = (_Float16)W[(size_t)K * nCols + N];
}

// ---------------------- WMMA GEMM kernels ----------------------------------

// h = X @ W + bias   (K=128, 128 output cols); one 16-row strip per wave.
__global__ void wmma_proj_kernel(const float* __restrict__ X,
                                 const _Float16* __restrict__ Wfrag,
                                 const float* __restrict__ bias, float* __restrict__ out,
                                 int nRows) {
    __shared__ __align__(32) _Float16 sm[8 * 2048];   // 32 KB
    stage_w<32768>(Wfrag, sm);
    const int lane = threadIdx.x & 31;
    const int wave = threadIdx.x >> 5;
    const int rowBase = (blockIdx.x * (blockDim.x >> 5) + wave) << 4;
    if (rowBase + 16 > nRows) return;
    v16h a0 = load_frag_a(X, HID, rowBase, 0,  lane);
    v16h a1 = load_frag_a(X, HID, rowBase, 32, lane);
    v16h a2 = load_frag_a(X, HID, rowBase, 64, lane);
    v16h a3 = load_frag_a(X, HID, rowBase, 96, lane);
#pragma unroll
    for (int ct = 0; ct < 8; ++ct) {
        const _Float16* bp = sm + ct * 2048 + (lane << 4);
        v8f acc = {};
        acc = wmma16(a0, *(const v16h*)(bp),        acc);
        acc = wmma16(a1, *(const v16h*)(bp + 512),  acc);
        acc = wmma16(a2, *(const v16h*)(bp + 1024), acc);
        acc = wmma16(a3, *(const v16h*)(bp + 1536), acc);
        const int col = (ct << 4) + (lane & 15);
        const float bv = bias[col];
        const int r0 = rowBase + ((lane >> 4) << 3);
#pragma unroll
        for (int r = 0; r < 8; ++r)
            out[(size_t)(r0 + r) * HID + col] = acc[r] + bv;
    }
}

// colsum[col] += sum_rows tanh(X @ kW + kb)  (semantic-attention score input)
__global__ void wmma_tanh_colsum_kernel(const float* __restrict__ X,
                                        const _Float16* __restrict__ Wfrag,
                                        const float* __restrict__ bias,
                                        float* __restrict__ colsum, int nRows) {
    __shared__ __align__(32) _Float16 sm[8 * 2048];   // 32 KB
    stage_w<32768>(Wfrag, sm);
    const int lane = threadIdx.x & 31;
    const int wave = threadIdx.x >> 5;
    const int rowBase = (blockIdx.x * (blockDim.x >> 5) + wave) << 4;
    if (rowBase + 16 > nRows) return;
    v16h a0 = load_frag_a(X, HID, rowBase, 0,  lane);
    v16h a1 = load_frag_a(X, HID, rowBase, 32, lane);
    v16h a2 = load_frag_a(X, HID, rowBase, 64, lane);
    v16h a3 = load_frag_a(X, HID, rowBase, 96, lane);
#pragma unroll
    for (int ct = 0; ct < 8; ++ct) {
        const _Float16* bp = sm + ct * 2048 + (lane << 4);
        v8f acc = {};
        acc = wmma16(a0, *(const v16h*)(bp),        acc);
        acc = wmma16(a1, *(const v16h*)(bp + 512),  acc);
        acc = wmma16(a2, *(const v16h*)(bp + 1024), acc);
        acc = wmma16(a3, *(const v16h*)(bp + 1536), acc);
        const int col = (ct << 4) + (lane & 15);
        const float bv = bias[col];
        float part = 0.f;
#pragma unroll
        for (int r = 0; r < 8; ++r) part += tanhf(acc[r] + bv);
        atomicAdd(&colsum[col], part);
    }
}

// out = (X*ga + gb) @ W + bias   (GraphNorm folded into A; 64 output cols)
__global__ void wmma_norm_out_kernel(const float* __restrict__ X,
                                     const float* __restrict__ ga, const float* __restrict__ gb,
                                     const _Float16* __restrict__ Wfrag,
                                     const float* __restrict__ bias,
                                     float* __restrict__ out, int nRows) {
    __shared__ __align__(32) _Float16 sm[4 * 2048];   // 16 KB
    stage_w<16384>(Wfrag, sm);
    const int lane = threadIdx.x & 31;
    const int wave = threadIdx.x >> 5;
    const int rowBase = (blockIdx.x * (blockDim.x >> 5) + wave) << 4;
    if (rowBase + 16 > nRows) return;
    v16h a0 = load_frag_a_affine(X, HID, rowBase, 0,  lane, ga, gb);
    v16h a1 = load_frag_a_affine(X, HID, rowBase, 32, lane, ga, gb);
    v16h a2 = load_frag_a_affine(X, HID, rowBase, 64, lane, ga, gb);
    v16h a3 = load_frag_a_affine(X, HID, rowBase, 96, lane, ga, gb);
#pragma unroll
    for (int ct = 0; ct < 4; ++ct) {
        const _Float16* bp = sm + ct * 2048 + (lane << 4);
        v8f acc = {};
        acc = wmma16(a0, *(const v16h*)(bp),        acc);
        acc = wmma16(a1, *(const v16h*)(bp + 512),  acc);
        acc = wmma16(a2, *(const v16h*)(bp + 1024), acc);
        acc = wmma16(a3, *(const v16h*)(bp + 1536), acc);
        const int col = (ct << 4) + (lane & 15);
        const float bv = bias[col];
        const int r0 = rowBase + ((lane >> 4) << 3);
#pragma unroll
        for (int r = 0; r < 8; ++r)
            out[(size_t)(r0 + r) * OUTC + col] = acc[r] + bv;
    }
}

// ---------------------- elementwise / edge kernels --------------------------

__global__ void fill_kernel(float* __restrict__ p, size_t n, float v) {
    size_t i = (size_t)blockIdx.x * blockDim.x + threadIdx.x;
    size_t st = (size_t)gridDim.x * blockDim.x;
    for (; i < n; i += st) p[i] = v;
}

__global__ void relu_kernel(float* __restrict__ p, size_t n) {
    size_t i = (size_t)blockIdx.x * blockDim.x + threadIdx.x;
    size_t st = (size_t)gridDim.x * blockDim.x;
    for (; i < n; i += st) p[i] = fmaxf(p[i], 0.f);
}

// al[n,h] = sum_d h[n, h*16+d] * att[h,d]
__global__ void att_dot_kernel(const float* __restrict__ hfeat, const float* __restrict__ att,
                               float* __restrict__ out, int n) {
    int i = blockIdx.x * blockDim.x + threadIdx.x;
    if (i >= n * NHEAD) return;
    int node = i >> 3, hd = i & 7;
    const float* hp = hfeat + (size_t)node * HID + hd * DHEAD;
    const float* ap = att + hd * DHEAD;
    float s = 0.f;
#pragma unroll
    for (int d = 0; d < DHEAD; ++d) s += hp[d] * ap[d];
    out[i] = s;
}

__device__ __forceinline__ float lrelu02(float x) { return x > 0.f ? x : 0.2f * x; }

// float atomic max via order-preserving int encoding
__device__ __forceinline__ void atomicMaxF32(float* addr, float v) {
    if (__float_as_int(v) >= 0)
        atomicMax((int*)addr, __float_as_int(v));
    else
        atomicMin((unsigned int*)addr, __float_as_uint(v));
}

__global__ void edge_max_kernel(const int* __restrict__ row, const int* __restrict__ col,
                                const float* __restrict__ als, const float* __restrict__ ald,
                                float* __restrict__ m, int E) {
    int e = blockIdx.x * blockDim.x + threadIdx.x;
    if (e >= E) return;
    int r = row[e], c = col[e];
#pragma unroll
    for (int h = 0; h < NHEAD; ++h) {
        float a = lrelu02(als[(size_t)r * NHEAD + h] + ald[(size_t)c * NHEAD + h]);
        atomicMaxF32(&m[(size_t)c * NHEAD + h], a);
    }
}

__global__ void edge_sum_kernel(const int* __restrict__ row, const int* __restrict__ col,
                                const float* __restrict__ als, const float* __restrict__ ald,
                                const float* __restrict__ m, float* __restrict__ ssum, int E) {
    int e = blockIdx.x * blockDim.x + threadIdx.x;
    if (e >= E) return;
    int r = row[e], c = col[e];
#pragma unroll
    for (int h = 0; h < NHEAD; ++h) {
        float a = lrelu02(als[(size_t)r * NHEAD + h] + ald[(size_t)c * NHEAD + h]);
        atomicAdd(&ssum[(size_t)c * NHEAD + h], __expf(a - m[(size_t)c * NHEAD + h]));
    }
}

__global__ void edge_scatter_kernel(const int* __restrict__ row, const int* __restrict__ col,
                                    const float* __restrict__ als, const float* __restrict__ ald,
                                    const float* __restrict__ m, const float* __restrict__ ssum,
                                    const float* __restrict__ hsrc, float* __restrict__ out,
                                    int E) {
    int e = blockIdx.x * blockDim.x + threadIdx.x;
    if (e >= E) return;
    int r = row[e], c = col[e];
    const float* hp = hsrc + (size_t)r * HID;
    __builtin_prefetch(hp, 0, 0);       // start pulling the L2-resident row
    float w[NHEAD];
#pragma unroll
    for (int h = 0; h < NHEAD; ++h) {
        float a = lrelu02(als[(size_t)r * NHEAD + h] + ald[(size_t)c * NHEAD + h]);
        float ex = __expf(a - m[(size_t)c * NHEAD + h]);
        w[h] = ex / (ssum[(size_t)c * NHEAD + h] + 1e-16f);
    }
    float* op = out + (size_t)c * HID;
#pragma unroll 4
    for (int q = 0; q < 32; ++q) {      // 32 x b128 source loads
        v4f hv = *(const v4f*)(hp + q * 4);
        float wv = w[q >> 2];
        atomicAdd(&op[q * 4 + 0], hv[0] * wv);
        atomicAdd(&op[q * 4 + 1], hv[1] * wv);
        atomicAdd(&op[q * 4 + 2], hv[2] * wv);
        atomicAdd(&op[q * 4 + 3], hv[3] * wv);
    }
}

// paper-side fusion: o0 = attn0*o0 + attn1*o1 (in place)
__global__ void fuse_kernel(float* __restrict__ o0, const float* __restrict__ o1,
                            const float* __restrict__ attn, size_t n) {
    float a0 = attn[0], a1 = attn[1];
    size_t i = (size_t)blockIdx.x * blockDim.x + threadIdx.x;
    size_t st = (size_t)gridDim.x * blockDim.x;
    for (; i < n; i += st) o0[i] = a0 * o0[i] + a1 * o1[i];
}

// semantic attention scores (papers: two metapaths) -> softmax weights
__global__ void semantic_attn_kernel(const float* __restrict__ colsum,
                                     const float* __restrict__ q, float invN,
                                     float* __restrict__ attn) {
    if (threadIdx.x == 0 && blockIdx.x == 0) {
        float s0 = 0.f, s1 = 0.f;
        for (int c = 0; c < HID; ++c) { s0 += q[c] * colsum[c]; s1 += q[c] * colsum[HID + c]; }
        s0 *= invN; s1 *= invN;
        float mx = fmaxf(s0, s1);
        float e0 = __expf(s0 - mx), e1 = __expf(s1 - mx);
        float d = e0 + e1;
        attn[0] = e0 / d; attn[1] = e1 / d;
    }
}

// GraphNorm column statistics: blockDim = 128 (one channel per thread)
__global__ void colsum_stats_kernel(const float* __restrict__ x, int nRows,
                                    float* __restrict__ sums) {
    const int c = threadIdx.x;
    const int r0 = blockIdx.x * 256;
    float acc = 0.f;
    for (int i = 0; i < 256; ++i) {
        int r = r0 + i;
        if (r < nRows) acc += x[(size_t)r * HID + c];
    }
    atomicAdd(&sums[c], acc);
}

__global__ void colvar_stats_kernel(const float* __restrict__ x, int nRows,
                                    const float* __restrict__ sums,
                                    const float* __restrict__ scale,
                                    float* __restrict__ vars) {
    const int c = threadIdx.x;
    const float sm = scale[c] * (sums[c] / (float)nRows);
    const int r0 = blockIdx.x * 256;
    float acc = 0.f;
    for (int i = 0; i < 256; ++i) {
        int r = r0 + i;
        if (r < nRows) { float d = x[(size_t)r * HID + c] - sm; acc += d * d; }
    }
    atomicAdd(&vars[c], acc);
}

// fold GraphNorm into per-channel affine: y = x*ga + gb
__global__ void norm_affine_kernel(const float* __restrict__ sums, const float* __restrict__ vars,
                                   const float* __restrict__ w, const float* __restrict__ b,
                                   const float* __restrict__ scale,
                                   float* __restrict__ ga, float* __restrict__ gb, int nRows) {
    int c = threadIdx.x;   // one block of 128
    float mu = sums[c] / (float)nRows;
    float var = vars[c] / (float)nRows;
    float g = w[c] * rsqrtf(var + 1e-5f);
    ga[c] = g;
    gb[c] = b[c] - scale[c] * mu * g;
}

// ---------------------------------------------------------------------------

extern "C" void kernel_launch(void* const* d_in, const int* in_sizes, int n_in,
                              void* d_out, int out_size, void* d_ws, size_t ws_size,
                              hipStream_t stream) {
    const float* x_a    = (const float*)d_in[0];
    const float* x_p    = (const float*)d_in[1];
    const float* Wpa    = (const float*)d_in[2];
    const float* bpa    = (const float*)d_in[3];
    const float* Wpp    = (const float*)d_in[4];
    const float* bpp    = (const float*)d_in[5];
    const float* as_a2p = (const float*)d_in[6];
    const float* ad_a2p = (const float*)d_in[7];
    const float* as_p2a = (const float*)d_in[8];
    const float* ad_p2a = (const float*)d_in[9];
    const float* as_p2p = (const float*)d_in[10];
    const float* ad_p2p = (const float*)d_in[11];
    const float* klW    = (const float*)d_in[12];
    const float* klb    = (const float*)d_in[13];
    const float* qv     = (const float*)d_in[14];
    const float* gw_a   = (const float*)d_in[15];
    const float* gbias_a= (const float*)d_in[16];
    const float* gs_a   = (const float*)d_in[17];
    const float* gw_p   = (const float*)d_in[18];
    const float* gbias_p= (const float*)d_in[19];
    const float* gs_p   = (const float*)d_in[20];
    const float* Woa    = (const float*)d_in[21];
    const float* boa    = (const float*)d_in[22];
    const float* Wop    = (const float*)d_in[23];
    const float* bop    = (const float*)d_in[24];
    const int*   e_a2p  = (const int*)d_in[25];
    const int*   e_p2a  = (const int*)d_in[26];
    const int*   e_p2p  = (const int*)d_in[27];

    const int NA = in_sizes[0] / HID;
    const int NP = in_sizes[1] / HID;
    const int E1 = in_sizes[25] / 2;   // a2p
    const int E2 = in_sizes[26] / 2;   // p2a
    const int E3 = in_sizes[27] / 2;   // p2p

    // ---- workspace layout (floats) ----
    float* ws = (float*)d_ws;
    size_t off = 0;
    auto alloc = [&](size_t n) { float* p = ws + off; off += n; return p; };
    float* h_a   = alloc((size_t)NA * HID);
    float* h_p   = alloc((size_t)NP * HID);
    float* o_a2p = alloc((size_t)NP * HID);   // fused in place -> paper output
    float* o_p2p = alloc((size_t)NP * HID);
    float* o_p2a = alloc((size_t)NA * HID);   // author output (attn == 1)
    const size_t oTotal = (size_t)(2 * NP + NA) * HID;
    float* als1 = alloc((size_t)NA * NHEAD);  // a2p src (authors)
    float* ald1 = alloc((size_t)NP * NHEAD);
    float* als2 = alloc((size_t)NP * NHEAD);  // p2p
    float* ald2 = alloc((size_t)NP * NHEAD);
    float* als3 = alloc((size_t)NP * NHEAD);  // p2a src (papers)
    float* ald3 = alloc((size_t)NA * NHEAD);
    float* m1 = alloc((size_t)NP * NHEAD);    // contiguous -inf region
    float* m2 = alloc((size_t)NP * NHEAD);
    float* m3 = alloc((size_t)NA * NHEAD);
    const size_t mTotal = (size_t)(2 * NP + NA) * NHEAD;
    float* s1 = alloc((size_t)NP * NHEAD);    // contiguous zero region from here
    float* s2 = alloc((size_t)NP * NHEAD);
    float* s3 = alloc((size_t)NA * NHEAD);
    float* csum  = alloc(2 * HID);
    float* sum_a = alloc(HID);
    float* var_a = alloc(HID);
    float* sum_p = alloc(HID);
    float* var_p = alloc(HID);
    float* attn  = alloc(4);                  // padded for alignment
    const size_t zTail = (size_t)(2 * NP + NA) * NHEAD + 2 * HID + 4 * HID + 4;
    float* ga_a = alloc(HID);
    float* gb_a = alloc(HID);
    float* ga_p = alloc(HID);
    float* gb_p = alloc(HID);
    // fragment-packed f16 weights (stored in float-sized slots)
    _Float16* wf_pa = (_Float16*)alloc(8192);   // 128-col: 16384 halfs
    _Float16* wf_pp = (_Float16*)alloc(8192);
    _Float16* wf_kl = (_Float16*)alloc(8192);
    _Float16* wf_oa = (_Float16*)alloc(4096);   // 64-col: 8192 halfs
    _Float16* wf_op = (_Float16*)alloc(4096);

    const dim3 b256(256), b128(128), bW(128);
    const int WPB = 4;                              // waves per block (wave32)
    const int rtA = NA / 16, rtP = NP / 16;
    auto cdiv = [](long long a, long long b) { return (unsigned)((a + b - 1) / b); };

    // ---- init accumulators ----
    fill_kernel<<<2048, b256, 0, stream>>>(o_a2p, oTotal, 0.f);
    fill_kernel<<<512,  b256, 0, stream>>>(m1, mTotal, -INFINITY);
    fill_kernel<<<512,  b256, 0, stream>>>(s1, zTail, 0.f);

    // ---- pack weights into f16 WMMA fragment order ----
    pack_wfrag_kernel<<<cdiv(16384, 256), b256, 0, stream>>>(Wpa, wf_pa, HID);
    pack_wfrag_kernel<<<cdiv(16384, 256), b256, 0, stream>>>(Wpp, wf_pp, HID);
    pack_wfrag_kernel<<<cdiv(16384, 256), b256, 0, stream>>>(klW, wf_kl, HID);
    pack_wfrag_kernel<<<cdiv(8192, 256),  b256, 0, stream>>>(Woa, wf_oa, OUTC);
    pack_wfrag_kernel<<<cdiv(8192, 256),  b256, 0, stream>>>(Wop, wf_op, OUTC);

    // ---- per-type projection GEMMs (TDM-staged B, WMMA f16 -> f32) ----
    wmma_proj_kernel<<<cdiv(rtA, WPB), bW, 0, stream>>>(x_a, wf_pa, bpa, h_a, NA);
    wmma_proj_kernel<<<cdiv(rtP, WPB), bW, 0, stream>>>(x_p, wf_pp, bpp, h_p, NP);

    // ---- per-node attention logits ----
    att_dot_kernel<<<cdiv((long long)NA * NHEAD, 256), b256, 0, stream>>>(h_a, as_a2p, als1, NA);
    att_dot_kernel<<<cdiv((long long)NP * NHEAD, 256), b256, 0, stream>>>(h_p, ad_a2p, ald1, NP);
    att_dot_kernel<<<cdiv((long long)NP * NHEAD, 256), b256, 0, stream>>>(h_p, as_p2p, als2, NP);
    att_dot_kernel<<<cdiv((long long)NP * NHEAD, 256), b256, 0, stream>>>(h_p, ad_p2p, ald2, NP);
    att_dot_kernel<<<cdiv((long long)NP * NHEAD, 256), b256, 0, stream>>>(h_p, as_p2a, als3, NP);
    att_dot_kernel<<<cdiv((long long)NA * NHEAD, 256), b256, 0, stream>>>(h_a, ad_p2a, ald3, NA);

    // ---- relation a2p: authors -> papers ----
    edge_max_kernel<<<cdiv(E1, 256), b256, 0, stream>>>(e_a2p, e_a2p + E1, als1, ald1, m1, E1);
    edge_sum_kernel<<<cdiv(E1, 256), b256, 0, stream>>>(e_a2p, e_a2p + E1, als1, ald1, m1, s1, E1);
    edge_scatter_kernel<<<cdiv(E1, 256), b256, 0, stream>>>(e_a2p, e_a2p + E1, als1, ald1,
                                                            m1, s1, h_a, o_a2p, E1);
    // ---- relation p2p ----
    edge_max_kernel<<<cdiv(E3, 256), b256, 0, stream>>>(e_p2p, e_p2p + E3, als2, ald2, m2, E3);
    edge_sum_kernel<<<cdiv(E3, 256), b256, 0, stream>>>(e_p2p, e_p2p + E3, als2, ald2, m2, s2, E3);
    edge_scatter_kernel<<<cdiv(E3, 256), b256, 0, stream>>>(e_p2p, e_p2p + E3, als2, ald2,
                                                            m2, s2, h_p, o_p2p, E3);
    // ---- relation p2a: papers -> authors ----
    edge_max_kernel<<<cdiv(E2, 256), b256, 0, stream>>>(e_p2a, e_p2a + E2, als3, ald3, m3, E2);
    edge_sum_kernel<<<cdiv(E2, 256), b256, 0, stream>>>(e_p2a, e_p2a + E2, als3, ald3, m3, s3, E2);
    edge_scatter_kernel<<<cdiv(E2, 256), b256, 0, stream>>>(e_p2a, e_p2a + E2, als3, ald3,
                                                            m3, s3, h_p, o_p2a, E2);

    // ---- ReLU on all relation outputs ----
    relu_kernel<<<2048, b256, 0, stream>>>(o_a2p, oTotal);

    // ---- semantic attention across paper metapaths (WMMA tanh column sums) ----
    wmma_tanh_colsum_kernel<<<cdiv(rtP, WPB), bW, 0, stream>>>(o_a2p, wf_kl, klb, csum, NP);
    wmma_tanh_colsum_kernel<<<cdiv(rtP, WPB), bW, 0, stream>>>(o_p2p, wf_kl, klb, csum + HID, NP);
    semantic_attn_kernel<<<1, 32, 0, stream>>>(csum, qv, 1.f / (float)NP, attn);
    fuse_kernel<<<1024, b256, 0, stream>>>(o_a2p, o_p2p, attn, (size_t)NP * HID);
    // authors: single metapath -> attn == 1, out_a = o_p2a directly

    // ---- GraphNorm statistics -> per-channel affine ----
    colsum_stats_kernel<<<cdiv(NA, 256), b128, 0, stream>>>(o_p2a, NA, sum_a);
    colvar_stats_kernel<<<cdiv(NA, 256), b128, 0, stream>>>(o_p2a, NA, sum_a, gs_a, var_a);
    norm_affine_kernel<<<1, 128, 0, stream>>>(sum_a, var_a, gw_a, gbias_a, gs_a, ga_a, gb_a, NA);
    colsum_stats_kernel<<<cdiv(NP, 256), b128, 0, stream>>>(o_a2p, NP, sum_p);
    colvar_stats_kernel<<<cdiv(NP, 256), b128, 0, stream>>>(o_a2p, NP, sum_p, gs_p, var_p);
    norm_affine_kernel<<<1, 128, 0, stream>>>(sum_p, var_p, gw_p, gbias_p, gs_p, ga_p, gb_p, NP);

    // ---- final (norm ∘ linear) GEMMs straight into d_out ----
    float* out_f = (float*)d_out;
    wmma_norm_out_kernel<<<cdiv(rtA, WPB), bW, 0, stream>>>(o_p2a, ga_a, gb_a, wf_oa, boa,
                                                            out_f, NA);
    wmma_norm_out_kernel<<<cdiv(rtP, WPB), bW, 0, stream>>>(o_a2p, ga_p, gb_p, wf_op, bop,
                                                            out_f + (size_t)NA * OUTC, NP);
}
